// BDGFunction_83519934038511
// MI455X (gfx1250) — 8-bit matrix operand order repaired and checked by reading, compiled, not run on hardware
//
#include <hip/hip_runtime.h>
#include <math.h>

typedef __attribute__((ext_vector_type(8))) int v8i;

#define BB 8
#define NN 192
#define MM 48
#define JEPS 1e-10
#define NSWEEP 12
#define SLICES 7

__global__ __launch_bounds__(1024) void k_build(const double* __restrict__ xin,
                                                const double* __restrict__ bre,
                                                const double* __restrict__ bim,
                                                double* __restrict__ H,
                                                double* __restrict__ Qm)
{
    int b = blockIdx.x;
    const double* br = bre + (size_t)b * NN * NN;
    const double* bi = bim + (size_t)b * NN * NN;
    double* Hb = H  + (size_t)b * NN * NN * 2;
    double* Qb = Qm + (size_t)b * NN * NN * 2;

    for (int e = threadIdx.x; e < NN * NN; e += blockDim.x) {
        Hb[2 * e]     = br[e];
        Hb[2 * e + 1] = bi[e];
        int r = e / NN, c = e % NN;
        Qb[2 * e]     = (r == c) ? 1.0 : 0.0;
        Qb[2 * e + 1] = 0.0;
    }
    __syncthreads();
    if (threadIdx.x < MM) {
        int j = threadIdx.x;
        double xv = xin[b * MM + j];
        int r0 = 4 * j, r1 = 4 * j + 1, c0 = 4 * j + 2, c1 = 4 * j + 3;
        auto setH = [&](int r, int c, double v) {
            Hb[2 * (r * NN + c)] = v; Hb[2 * (r * NN + c) + 1] = 0.0;
        };
        setH(r0, c0, 0.0);  setH(r0, c1,  xv);
        setH(r1, c0, -xv);  setH(r1, c1, 0.0);
        setH(c0, r0, 0.0);  setH(c0, r1, -xv);
        setH(c1, r0,  xv);  setH(c1, r1, 0.0);
    }
}

__global__ __launch_bounds__(1024) void k_jacobi(double* __restrict__ H,
                                                 double* __restrict__ Qm,
                                                 double* __restrict__ Lw)
{
    int b = blockIdx.x;
    double* A  = H  + (size_t)b * NN * NN * 2;
    double* Qb = Qm + (size_t)b * NN * NN * 2;

    __shared__ double cs_c[NN / 2], cs_sr[NN / 2], cs_si[NN / 2];
    __shared__ int    pr_p[NN / 2], pr_q[NN / 2];
    const int NP = NN / 2;
    const int NR = NN - 1;

    for (int sweep = 0; sweep < NSWEEP; ++sweep) {
        for (int r = 0; r < NR; ++r) {
            if (threadIdx.x < NP) {
                int k = threadIdx.x, p, q;
                if (k == 0) { p = NN - 1; q = r % NR; }
                else        { p = (r + k) % NR; q = (r - k + 2 * NR) % NR; }
                double app = A[2 * (p * NN + p)];
                double aqq = A[2 * (q * NN + q)];
                double gr  = A[2 * (p * NN + q)];
                double gi  = A[2 * (p * NN + q) + 1];
                double g   = sqrt(gr * gr + gi * gi);
                double c = 1.0, sr = 0.0, si = 0.0;
                if (g > 1e-150) {
                    double d    = app - aqq;
                    double tden = fabs(d) + sqrt(d * d + 4.0 * g * g);
                    double tt   = copysign(2.0 * g / tden, d);
                    c = 1.0 / sqrt(1.0 + tt * tt);
                    double sig = tt * c;
                    sr =  sig * gr / g;
                    si = -sig * gi / g;
                }
                pr_p[k] = p; pr_q[k] = q;
                cs_c[k] = c; cs_sr[k] = sr; cs_si[k] = si;
            }
            __syncthreads();
            for (int e = threadIdx.x; e < NP * NN * 2; e += blockDim.x) {
                int k = e % NP, rem = e / NP, i = rem % NN, mat = rem / NN;
                double* Mx = mat ? Qb : A;
                int p = pr_p[k], q = pr_q[k];
                double c = cs_c[k], sr = cs_sr[k], si = cs_si[k];
                double Xr = Mx[2 * (i * NN + p)], Xi = Mx[2 * (i * NN + p) + 1];
                double Yr = Mx[2 * (i * NN + q)], Yi = Mx[2 * (i * NN + q) + 1];
                double npr = c * Xr + (sr * Yr - si * Yi);
                double npi = c * Xi + (sr * Yi + si * Yr);
                double nqr = -(sr * Xr + si * Xi) + c * Yr;
                double nqi = -(sr * Xi - si * Xr) + c * Yi;
                Mx[2 * (i * NN + p)] = npr; Mx[2 * (i * NN + p) + 1] = npi;
                Mx[2 * (i * NN + q)] = nqr; Mx[2 * (i * NN + q) + 1] = nqi;
            }
            __syncthreads();
            for (int e = threadIdx.x; e < NP * NN; e += blockDim.x) {
                int k = e % NP, i = e / NP;
                int p = pr_p[k], q = pr_q[k];
                double c = cs_c[k], sr = cs_sr[k], si = cs_si[k];
                double Xr = A[2 * (p * NN + i)], Xi = A[2 * (p * NN + i) + 1];
                double Yr = A[2 * (q * NN + i)], Yi = A[2 * (q * NN + i) + 1];
                double npr = c * Xr + (sr * Yr + si * Yi);
                double npi = c * Xi + (sr * Yi - si * Yr);
                double nqr = -(sr * Xr - si * Xi) + c * Yr;
                double nqi = -(sr * Xi + si * Xr) + c * Yi;
                A[2 * (p * NN + i)] = npr; A[2 * (p * NN + i) + 1] = npi;
                A[2 * (q * NN + i)] = nqr; A[2 * (q * NN + i) + 1] = nqi;
            }
            __syncthreads();
        }
    }
    for (int n = threadIdx.x; n < NN; n += blockDim.x)
        Lw[b * NN + n] = A[2 * (n * NN + n)];
}

__global__ __launch_bounds__(256) void k_post(const double* __restrict__ xin,
                                              const double* __restrict__ beta_p,
                                              const double* __restrict__ pot,
                                              const double* __restrict__ Qm,
                                              const double* __restrict__ Lw,
                                              double* __restrict__ uw,
                                              double* __restrict__ vw,
                                              double* __restrict__ w3,
                                              double* __restrict__ outv,
                                              signed char* __restrict__ bsl,
                                              int* __restrict__ sBw)
{
    int b = blockIdx.x;
    double beta = beta_p[0];
    const double* Qb = Qm + (size_t)b * NN * NN * 2;
    __shared__ double t_s[NN], dt_s[NN];

    for (int n = threadIdx.x; n < NN; n += blockDim.x) {
        double L = Lw[b * NN + n];
        double t = tanh(0.5 * beta * L);
        t_s[n] = t;
        dt_s[n] = 0.5 * beta * (1.0 - t * t);
    }
    __syncthreads();

    for (int e = threadIdx.x; e < MM * NN; e += blockDim.x) {
        int i = e / NN, n = e % NN;
        double ur = Qb[2 * ((4 * i) * NN + n)],     ui = Qb[2 * ((4 * i) * NN + n) + 1];
        double vr = Qb[2 * ((4 * i + 3) * NN + n)], vi = Qb[2 * ((4 * i + 3) * NN + n) + 1];
        size_t o = ((size_t)(b * MM + i) * NN + n) * 2;
        uw[o] = ur; uw[o + 1] = ui;
        vw[o] = vr; vw[o + 1] = vi;
        double pr = ur * vr + ui * vi;
        double pi = ui * vr - ur * vi;
        w3[o] = pr * dt_s[n]; w3[o + 1] = pi * dt_s[n];
    }
    __syncthreads();

    if (threadIdx.x < MM) {
        int i = threadIdx.x;
        double fr = 0, fi = 0;
        for (int n = 0; n < NN; ++n) {
            size_t o = ((size_t)(b * MM + i) * NN + n) * 2;
            double ur = uw[o], ui = uw[o + 1], vr = vw[o], vi = vw[o + 1];
            fr += (ur * vr + ui * vi) * t_s[n];
            fi += (ui * vr - ur * vi) * t_s[n];
        }
        double s = 0.5 * pot[i];
        outv[(b * MM + i) * 2]     = s * fr - xin[b * MM + i];
        outv[(b * MM + i) * 2 + 1] = s * fi;
    }
    __syncthreads();

    for (int row = threadIdx.x; row < 4 * MM; row += blockDim.x) {
        int tp = row / MM, i = row % MM;
        double mx = 0.0;
        for (int n = 0; n < NN; ++n) {
            size_t o = ((size_t)(b * MM + i) * NN + n) * 2;
            double val = (tp == 0) ?  vw[o]     * t_s[n]
                       : (tp == 1) ? -vw[o + 1] * t_s[n]
                       : (tp == 2) ?  uw[o]     * t_s[n]
                                   : -uw[o + 1] * t_s[n];
            mx = fmax(mx, fabs(val));
        }
        int sc = (mx > 0.0) ? (ilogb(mx) + 2) : 0;
        sBw[b * 192 + row] = sc;
        for (int n = 0; n < NN; ++n) {
            size_t o = ((size_t)(b * MM + i) * NN + n) * 2;
            double val = (tp == 0) ?  vw[o]     * t_s[n]
                       : (tp == 1) ? -vw[o + 1] * t_s[n]
                       : (tp == 2) ?  uw[o]     * t_s[n]
                                   : -uw[o + 1] * t_s[n];
            double r = ldexp(val, -sc);
            for (int q = 0; q < SLICES; ++q) {
                r *= 128.0;
                double dg = rint(r);
                r -= dg;
                bsl[((size_t)(b * SLICES + q) * 192 + row) * 192 + n] = (signed char)(int)dg;
            }
        }
    }
}

__device__ __forceinline__ void compC(const double* r0s, const double* r1s,
                                      const double* c0s, const double* c1s,
                                      const double* Ls, int m, int n,
                                      double* cr, double* ci)
{
    double ar = r0s[2 * m], ai = -r0s[2 * m + 1];
    double brr = c1s[2 * n], bii = c1s[2 * n + 1];
    double pr = ar * brr - ai * bii, pi = ar * bii + ai * brr;
    double crr = r1s[2 * m], cii = -r1s[2 * m + 1];
    double drr = c0s[2 * n], dii = c0s[2 * n + 1];
    pr -= (crr * drr - cii * dii);
    pi -= (crr * dii + cii * drr);
    double er = r0s[2 * n], ei = -r0s[2 * n + 1];
    double fr = c1s[2 * m], fi = c1s[2 * m + 1];
    double qr = er * fr - ei * fi, qi = er * fi + ei * fr;
    double gr = r1s[2 * n], gi = -r1s[2 * n + 1];
    double hr = c0s[2 * m], hi = c0s[2 * m + 1];
    qr -= (gr * hr - gi * hi);
    qi -= (gr * hi + gi * hr);
    double mr = pr + qr, mi = pi - qi;
    double D = Ls[n] - Ls[m];
    if (fabs(D) > JEPS) { double inv = 1.0 / D; *cr = mr * inv; *ci = mi * inv; }
    else                { *cr = 0.0; *ci = 0.0; }
}

__global__ __launch_bounds__(256) void k_main(const double* __restrict__ Qm,
                                              const double* __restrict__ Lw,
                                              const double* __restrict__ uw,
                                              const double* __restrict__ vw,
                                              const double* __restrict__ w3,
                                              const double* __restrict__ pot,
                                              const signed char* __restrict__ bsl,
                                              const int* __restrict__ sBw,
                                              double* __restrict__ outv)
{
    int b = blockIdx.x / MM;
    int j = blockIdx.x % MM;
    int tid = threadIdx.x;
    int lane = tid & 31;
    int wv = tid >> 5;

    __shared__ double r0s[NN * 2], r1s[NN * 2], c0s[NN * 2], c1s[NN * 2];
    __shared__ double Ls[NN], dEs[NN];
    __shared__ __align__(16) signed char Asl[2 * SLICES * 16 * NN];
    __shared__ int sA[32];
    __shared__ __align__(16) signed char Bsl[SLICES * 4 * 16 * NN];
    __shared__ int sB[64];
    __shared__ double Zb[8 * 16 * 16];
    __shared__ double redmax[32 * 8];
    __shared__ double t1r[MM], t1i[MM], i2r[MM], i2i[MM], t3r[MM], t3i[MM];

    const double* Qb = Qm + (size_t)b * NN * NN * 2;
    for (int n = tid; n < NN; n += 256) {
        r0s[2 * n] = Qb[2 * ((4 * j) * NN + n)];     r0s[2 * n + 1] = Qb[2 * ((4 * j) * NN + n) + 1];
        r1s[2 * n] = Qb[2 * ((4 * j + 1) * NN + n)]; r1s[2 * n + 1] = Qb[2 * ((4 * j + 1) * NN + n) + 1];
        c0s[2 * n] = Qb[2 * ((4 * j + 2) * NN + n)]; c0s[2 * n + 1] = Qb[2 * ((4 * j + 2) * NN + n) + 1];
        c1s[2 * n] = Qb[2 * ((4 * j + 3) * NN + n)]; c1s[2 * n + 1] = Qb[2 * ((4 * j + 3) * NN + n) + 1];
        Ls[n] = Lw[b * NN + n];
    }
    if (tid < MM) { t1r[tid] = t1i[tid] = i2r[tid] = i2i[tid] = 0.0; }
    __syncthreads();

    for (int n = tid; n < NN; n += 256) {
        double pr = (r0s[2 * n] * c1s[2 * n] + r0s[2 * n + 1] * c1s[2 * n + 1])
                  - (r1s[2 * n] * c0s[2 * n] + r1s[2 * n + 1] * c0s[2 * n + 1]);
        dEs[n] = 2.0 * pr;
    }
    __syncthreads();

    if (tid < MM) {
        int i = tid;
        double ar = 0, ai = 0;
        for (int n = 0; n < NN; ++n) {
            size_t o = ((size_t)(b * MM + i) * NN + n) * 2;
            ar += w3[o] * dEs[n];
            ai += w3[o + 1] * dEs[n];
        }
        t3r[i] = ar; t3i[i] = ai;
    }
    __syncthreads();

    for (int ib = 0; ib < 3; ++ib) {
        for (int idx = tid; idx < SLICES * 4 * NN; idx += 256) {
            int chunk = idx / NN;
            int off16 = idx % NN;
            int q = chunk >> 2, t = chunk & 3;
            unsigned long long src = (unsigned long long)(const void*)
                (bsl + (((size_t)(b * SLICES + q) * 192 + t * 48 + ib * 16) * 192) + (size_t)off16 * 16);
            unsigned lds = (unsigned)(unsigned long long)(const void*)
                (Bsl + (size_t)chunk * 16 * NN + (size_t)off16 * 16);
            asm volatile("global_load_async_to_lds_b128 %0, %1, off"
                         :: "v"(lds), "v"(src) : "memory");
        }
        asm volatile("s_wait_asynccnt 0" ::: "memory");
        if (tid < 64) sB[tid] = sBw[b * 192 + (tid / 16) * 48 + ib * 16 + (tid % 16)];
        __syncthreads();

        for (int mt = 0; mt < 12; ++mt) {
            int arow = tid >> 3;
            int sub  = tid & 7;
            int part = arow >> 4;
            int ml   = arow & 15;
            int mg   = mt * 16 + ml;
            double mx = 0.0;
            for (int n = sub * 24; n < sub * 24 + 24; ++n) {
                double crv, civ;
                compC(r0s, r1s, c0s, c1s, Ls, mg, n, &crv, &civ);
                mx = fmax(mx, fabs(part ? civ : crv));
            }
            redmax[arow * 8 + sub] = mx;
            __syncthreads();
            if (sub == 0) {
                double m2 = redmax[arow * 8];
                for (int z = 1; z < 8; ++z) m2 = fmax(m2, redmax[arow * 8 + z]);
                sA[arow] = (m2 > 0.0) ? (ilogb(m2) + 2) : 0;
            }
            __syncthreads();
            {
                int sc = sA[arow];
                for (int n = sub * 24; n < sub * 24 + 24; ++n) {
                    double crv, civ;
                    compC(r0s, r1s, c0s, c1s, Ls, mg, n, &crv, &civ);
                    double r = ldexp(part ? civ : crv, -sc);
                    for (int q = 0; q < SLICES; ++q) {
                        r *= 128.0;
                        double dg = rint(r);
                        r -= dg;
                        Asl[((part * SLICES + q) * 16 + ml) * NN + n] = (signed char)(int)dg;
                    }
                }
            }
            __syncthreads();

            {
                int prt = wv & 1;
                int ntile = wv >> 1;
                int mrow = lane & 15;
                int khA = (lane < 16) ? 0 : 16;
                int ncol = lane & 15;
                int khB = (lane < 16) ? 0 : 16;
                double facc[8];
                #pragma unroll
                for (int z = 0; z < 8; ++z) facc[z] = 0.0;
                int sBv = sB[ntile * 16 + ncol];

                for (int p = 0; p < SLICES; ++p) {
                    for (int q = 0; q + p < SLICES; ++q) {
                        v8i acc;
                        #pragma unroll
                        for (int z = 0; z < 8; ++z) acc[z] = 0;
                        #pragma unroll
                        for (int kt = 0; kt < 3; ++kt) {
                            v8i Af, Bf;
                            const signed char* ap = &Asl[((prt * SLICES + p) * 16 + mrow) * NN];
                            #pragma unroll
                            for (int jj = 0; jj < 4; ++jj) {
                                int k0 = kt * 64 + (jj & 1) * 8 + (jj >> 1) * 32 + khA;
                                Af[2 * jj]     = *(const int*)(ap + k0);
                                Af[2 * jj + 1] = *(const int*)(ap + k0 + 4);
                            }
                            const signed char* bp = &Bsl[((q * 4 + ntile) * 16 + ncol) * NN];
                            int k0b = kt * 64 + khB;
                            int4 blo = *(const int4*)(bp + k0b);
                            int4 bhi = *(const int4*)(bp + k0b + 32);
                            Bf[0] = blo.x; Bf[1] = blo.y; Bf[2] = blo.z; Bf[3] = blo.w;
                            Bf[4] = bhi.x; Bf[5] = bhi.y; Bf[6] = bhi.z; Bf[7] = bhi.w;
                            acc = __builtin_amdgcn_wmma_i32_16x16x64_iu8(true, Af, true, Bf,
                                                                         acc, false, false);
                        }
                        int sh = -7 * (p + q + 2);
                        #pragma unroll
                        for (int z = 0; z < 8; ++z) {
                            int Mr = z + ((lane < 16) ? 0 : 8);
                            facc[z] += ldexp((double)acc[z], sA[prt * 16 + Mr] + sBv + sh);
                        }
                    }
                }
                #pragma unroll
                for (int z = 0; z < 8; ++z) {
                    int Mr = z + ((lane < 16) ? 0 : 8);
                    Zb[(wv * 16 + Mr) * 16 + ncol] = facc[z];
                }
            }
            __syncthreads();

            if (tid < 16) {
                int il = tid, i = ib * 16 + il;
                double a1r = 0, a1i = 0, a2r = 0, a2i = 0;
                for (int mlc = 0; mlc < 16; ++mlc) {
                    int m = mt * 16 + mlc;
                    double zrg = Zb[((0) * 16 + mlc) * 16 + il] - Zb[((3) * 16 + mlc) * 16 + il];
                    double zig = Zb[((2) * 16 + mlc) * 16 + il] + Zb[((1) * 16 + mlc) * 16 + il];
                    double zrh = Zb[((4) * 16 + mlc) * 16 + il] - Zb[((7) * 16 + mlc) * 16 + il];
                    double zih = Zb[((6) * 16 + mlc) * 16 + il] + Zb[((5) * 16 + mlc) * 16 + il];
                    size_t o = ((size_t)(b * MM + i) * NN + m) * 2;
                    double ur = uw[o], ui = uw[o + 1];
                    double vr = vw[o], vi = vw[o + 1];
                    a1r += ur * zrg - ui * zig;
                    a1i += ur * zig + ui * zrg;
                    a2r += vr * zrh - vi * zih;
                    a2i += vr * zih + vi * zrh;
                }
                t1r[i] += a1r; t1i[i] += a1i;
                i2r[i] += a2r; i2i[i] += a2i;
            }
            __syncthreads();
        }
    }

    if (tid < MM) {
        int i = tid;
        double sr = t1r[i] + i2r[i] + t3r[i];
        double si = t1i[i] - i2i[i] + t3i[i];
        double sc = 0.5 * pot[i];
        size_t o = (size_t)BB * MM * 2 + ((size_t)(b * MM + i) * MM + j) * 2;
        outv[o]     = sc * sr - ((i == j) ? 1.0 : 0.0);
        outv[o + 1] = sc * si;
    }
}

extern "C" void kernel_launch(void* const* d_in, const int* in_sizes, int n_in,
                              void* d_out, int out_size, void* d_ws, size_t ws_size,
                              hipStream_t stream)
{
    (void)in_sizes; (void)n_in; (void)out_size; (void)ws_size;
    const double* x    = (const double*)d_in[0];
    const double* bre  = (const double*)d_in[1];
    const double* bim  = (const double*)d_in[2];
    const double* beta = (const double*)d_in[3];
    const double* pot  = (const double*)d_in[5];
    double* out = (double*)d_out;

    double* ws = (double*)d_ws;
    size_t o = 0;
    double* H  = ws + o; o += (size_t)BB * NN * NN * 2;
    double* Qm = ws + o; o += (size_t)BB * NN * NN * 2;
    double* Lw = ws + o; o += (size_t)BB * NN;
    double* uw = ws + o; o += (size_t)BB * MM * NN * 2;
    double* vw = ws + o; o += (size_t)BB * MM * NN * 2;
    double* w3 = ws + o; o += (size_t)BB * MM * NN * 2;
    int* sBw = (int*)(ws + o);
    signed char* bsl = (signed char*)(sBw + BB * 4 * MM);

    k_build <<<BB, 1024, 0, stream>>>(x, bre, bim, H, Qm);
    k_jacobi<<<BB, 1024, 0, stream>>>(H, Qm, Lw);
    k_post  <<<BB, 256, 0, stream>>>(x, beta, pot, Qm, Lw, uw, vw, w3, out, bsl, sBw);
    k_main  <<<BB * MM, 256, 0, stream>>>(Qm, Lw, uw, vw, w3, pot, bsl, sBw, out);
}
